// DGQA_20315195310699
// MI455X (gfx1250) — compile-verified, matrix operand order checked by reading, not run on hardware
//
#include <hip/hip_runtime.h>
#include <stdint.h>
#include <math.h>

// Problem constants (from reference): B,P,DIM,H,HKV = 4,2048,1024,16,8 ; D=64
#define CB    4
#define CP    2048
#define CDIM  1024
#define CH    16
#define CHKV  8
#define CD    64
#define CKV   (CHKV * CD)   /* 512 */

typedef __attribute__((ext_vector_type(16))) __bf16 v16bf;
typedef __attribute__((ext_vector_type(8)))  float  v8f;
typedef __attribute__((ext_vector_type(4)))  unsigned int u32x4;
typedef __attribute__((ext_vector_type(8)))  unsigned int u32x8;

static __device__ __forceinline__ unsigned short f2bf(float f) {
  union { float f; unsigned u; } x; x.f = f;
  unsigned r = x.u + 0x7FFFu + ((x.u >> 16) & 1u);   // round-to-nearest-even
  return (unsigned short)(r >> 16);
}
static __device__ __forceinline__ float bf2f(unsigned short h) {
  union { unsigned u; float f; } x; x.u = ((unsigned)h) << 16;
  return x.f;
}

// A/B fragment loader for v_wmma_*_16x16x32_bf16 (wave32), from global memory.
// 16-bit A layout (ISA 7.12.2): lanes 0-15 hold rows M=0..15 with K {0..7,16..23},
// lanes 16-31 hold the same rows with K {8..15,24..31}; two b128 loads per lane.
static __device__ __forceinline__ v16bf load_frag16(const unsigned short* base,
                                                    int row0, int ld, int k0) {
  int lane = threadIdx.x & 31;
  int m    = lane & 15;
  int sel  = lane >> 4;
  const unsigned short* p = base + (size_t)(row0 + m) * (size_t)ld + k0 + sel * 8;
  union { uint4 q[2]; v16bf v; } u;
  u.q[0] = *(const uint4*)(p);        // K chunk +0 / +8
  u.q[1] = *(const uint4*)(p + 16);   // K chunk +16 / +24
  return u.v;
}

static __device__ __forceinline__ v8f wmma_bf16(v16bf a, v16bf b, v8f c) {
  return __builtin_amdgcn_wmma_f32_16x16x32_bf16(false, a, false, b, (short)0, c,
                                                 false, false);
}

// ---------------------------------------------------------------------------
// Tensor Data Mover: build a 2D D# (groups 0/1, ISA 8.3/8.4) and issue
// TENSOR_LOAD_TO_LDS.  data_size=2B, tile = tile_d1 rows x tile_d0 elements,
// row stride = stride0 elements.  The destination LDS pointer is passed both
// as the D# lds_addr (low 32 flat bits == LDS byte offset) and as an asm
// operand so the LDS object escapes and the "memory" clobber covers it —
// this is what keeps the later ds_load fragment reads alive.
// Issued per-wave; tracked with TENSORcnt.
// ---------------------------------------------------------------------------
static __device__ __forceinline__ void tdm_load_2d(
    unsigned short* lds_ptr, unsigned long long gaddr, unsigned tensor_d0,
    unsigned tensor_d1, unsigned tile_d0, unsigned tile_d1, unsigned stride0) {
  unsigned lds_off = (unsigned)(uintptr_t)(void*)lds_ptr;  // LDS byte offset
  u32x4 g0;
  g0.x = 1u;                                         // count=1, user descriptor
  g0.y = lds_off;                                    // lds_addr (bytes)
  g0.z = (unsigned)(gaddr & 0xFFFFFFFFu);            // global_addr[31:0]
  g0.w = (unsigned)((gaddr >> 32) & 0x1FFFFFFu) | (2u << 30);  // [56:32] | type=2
  u32x8 g1;
  g1[0] = 1u << 16;                                  // data_size = 2 bytes
  g1[1] = (tensor_d0 & 0xFFFFu) << 16;               // tensor_dim0[15:0]
  g1[2] = ((tensor_d0 >> 16) & 0xFFFFu) | ((tensor_d1 & 0xFFFFu) << 16);
  g1[3] = ((tensor_d1 >> 16) & 0xFFFFu) | ((tile_d0 & 0xFFFFu) << 16);
  g1[4] = tile_d1 & 0xFFFFu;                         // tile_dim1 (tile_dim2=0)
  g1[5] = stride0;                                   // tensor_dim0_stride[31:0]
  g1[6] = 0u;                                        // stride hi / dim1_stride lo
  g1[7] = 0u;
  asm volatile("tensor_load_to_lds %0, %1"
               :: "s"(g0), "s"(g1), "r"(lds_ptr) : "memory");
}

// Fragment loader from an LDS tile with leading dimension 32 (one K-step).
// Non-volatile: address-space inference turns these into ds_load_b128; the
// loads survive because the tile pointer escapes into the TDM asm above.
static __device__ __forceinline__ v16bf frag_lds32(const unsigned short* s,
                                                   int row0) {
  int lane = threadIdx.x & 31;
  int m    = lane & 15;
  int sel  = lane >> 4;
  const unsigned short* p = s + (row0 + m) * 32 + sel * 8;
  union { u32x4 q[2]; v16bf v; } u;
  u.q[0] = *(const u32x4*)(p);
  u.q[1] = *(const u32x4*)(p + 16);
  return u.v;
}

// ---------------------------------------------------------------------------
// Elementwise conversion / transpose helpers
// ---------------------------------------------------------------------------
__global__ void dgqa_cvt_bf16(const float* __restrict__ in,
                              unsigned short* __restrict__ out, int n) {
  for (int i = blockIdx.x * blockDim.x + threadIdx.x; i < n;
       i += gridDim.x * blockDim.x)
    out[i] = f2bf(in[i]);
}

// W: K x N (row-major, f32) -> Wt: N x K (row-major, bf16)
__global__ void dgqa_transpose_cvt(const float* __restrict__ W,
                                   unsigned short* __restrict__ Wt,
                                   int K, int N) {
  int n = K * N;
  for (int i = blockIdx.x * blockDim.x + threadIdx.x; i < n;
       i += gridDim.x * blockDim.x) {
    int k = i / N, c = i - k * N;
    Wt[(size_t)c * K + k] = f2bf(W[i]);
  }
}

// v: (B*P) x 512 bf16 -> vT: (B*HKV*64) x P bf16   (V transposed per kv-head)
__global__ void dgqa_transpose_v(const unsigned short* __restrict__ vb,
                                 unsigned short* __restrict__ vT) {
  int n = CB * CP * CKV;
  for (int i = blockIdx.x * blockDim.x + threadIdx.x; i < n;
       i += gridDim.x * blockDim.x) {
    int row = i >> 9;          // b*P + p
    int col = i & 511;         // hkv*64 + d
    int b = row >> 11, p = row & (CP - 1);
    int hkv = col >> 6, d = col & 63;
    vT[((size_t)(b * CHKV + hkv) * CD + d) * CP + p] = vb[i];
  }
}

// ---------------------------------------------------------------------------
// bf16 WMMA GEMM with TDM double-buffered LDS staging.
// C(MxN) = alpha * A(MxK) * B(KxN) [+ bias], B supplied as Bt = N x K.
// block = 256 threads = 8 waves (2x4), wave tile 64x64 (4x4 16x16 frags).
// Per K-step: wave0 issues two TENSOR_LOAD_TO_LDS (A: 128x32, B: 256x32) for
// the next stage, everyone computes the current stage from LDS, wave0 drains
// TENSORcnt, block barrier, buffers swap.  LDS = 2*(8KB + 16KB) = 48KB.
// ---------------------------------------------------------------------------
#define GA_ELEMS (128 * 32)     /* A stage tile, elements  */
#define GB_ELEMS (256 * 32)     /* B stage tile, elements  */

__global__ __launch_bounds__(256) void dgqa_gemm_bf16(
    const unsigned short* __restrict__ A, const unsigned short* __restrict__ Bt,
    int M, int N, int K, float alpha, const float* __restrict__ bias,
    unsigned short* __restrict__ outb, float* __restrict__ outf) {
  __shared__ unsigned short smem[2 * GA_ELEMS + 2 * GB_ELEMS];

  int wave = threadIdx.x >> 5;
  int lane = threadIdx.x & 31;
  int wm = wave >> 2, wn = wave & 3;
  int rb = blockIdx.y * 128;            // block row base
  int cb = blockIdx.x * 256;            // block col base

  v8f c[4][4];
#pragma unroll
  for (int i = 0; i < 4; ++i)
#pragma unroll
    for (int j = 0; j < 4; ++j)
#pragma unroll
      for (int r = 0; r < 8; ++r) c[i][j][r] = 0.0f;

  // Wave 0 issues both tile DMAs for K-chunk k0 into stage s.
  auto issue = [&](int k0, int s) {
    if (threadIdx.x < 32) {
      unsigned long long ga = (unsigned long long)(uintptr_t)A +
                              ((unsigned long long)rb * K + k0) * 2ull;
      tdm_load_2d(&smem[s * GA_ELEMS], ga, (unsigned)K, (unsigned)M, 32u, 128u,
                  (unsigned)K);
      unsigned long long gb = (unsigned long long)(uintptr_t)Bt +
                              ((unsigned long long)cb * K + k0) * 2ull;
      tdm_load_2d(&smem[2 * GA_ELEMS + s * GB_ELEMS], gb, (unsigned)K,
                  (unsigned)N, 32u, 256u, (unsigned)K);
    }
  };

  int cur = 0;
  issue(0, 0);
  if (threadIdx.x < 32) __builtin_amdgcn_s_wait_tensorcnt(0);
  __syncthreads();

  for (int k0 = 0; k0 < K; k0 += 32) {
    if (k0 + 32 < K) issue(k0 + 32, cur ^ 1);   // overlap DMA with WMMA

    const unsigned short* As = &smem[cur * GA_ELEMS];
    const unsigned short* Bs = &smem[2 * GA_ELEMS + cur * GB_ELEMS];
    v16bf a[4], b[4];
#pragma unroll
    for (int i = 0; i < 4; ++i) a[i] = frag_lds32(As, wm * 64 + 16 * i);
#pragma unroll
    for (int j = 0; j < 4; ++j) b[j] = frag_lds32(Bs, wn * 64 + 16 * j);
#pragma unroll
    for (int i = 0; i < 4; ++i)
#pragma unroll
      for (int j = 0; j < 4; ++j) c[i][j] = wmma_bf16(a[i], b[j], c[i][j]);

    if (threadIdx.x < 32) __builtin_amdgcn_s_wait_tensorcnt(0);
    __syncthreads();
    cur ^= 1;
  }

  int m = lane & 15, sel = lane >> 4;
  int row0 = rb + wm * 64, col0 = cb + wn * 64;
#pragma unroll
  for (int i = 0; i < 4; ++i)
#pragma unroll
    for (int j = 0; j < 4; ++j)
#pragma unroll
      for (int r = 0; r < 8; ++r) {
        int row = row0 + 16 * i + r + sel * 8;
        int col = col0 + 16 * j + m;
        float v = c[i][j][r] * alpha;
        if (outf) {
          outf[(size_t)row * N + col] = v + (bias ? bias[col] : 0.0f);
        } else {
          outb[(size_t)row * N + col] = f2bf(v);
        }
      }
}

// ---------------------------------------------------------------------------
// k magnitudes: per (b,hkv) sum of squares over (P,D)
// ---------------------------------------------------------------------------
__global__ __launch_bounds__(256) void dgqa_ksumsq(
    const unsigned short* __restrict__ kb, float* __restrict__ out) {
  int blk = blockIdx.x;               // b*HKV + hkv
  int b = blk / CHKV, hkv = blk - b * CHKV;
  const unsigned short* base = kb + (size_t)b * CP * CKV + hkv * CD;
  float s = 0.0f;
  for (int i = threadIdx.x; i < CP * CD; i += 256) {
    int p = i >> 6, d = i & 63;
    float v = bf2f(base[(size_t)p * CKV + d]);
    s += v * v;
  }
  __shared__ float red[256];
  red[threadIdx.x] = s;
  __syncthreads();
  for (int off = 128; off; off >>= 1) {
    if ((int)threadIdx.x < off) red[threadIdx.x] += red[threadIdx.x + off];
    __syncthreads();
  }
  if (threadIdx.x == 0) out[blk] = red[0];
}

// ---------------------------------------------------------------------------
// ratios + kv_id (exact reference semantics, tiny -> single thread)
// ---------------------------------------------------------------------------
__global__ void dgqa_ratios(const float* __restrict__ ksumsq,
                            const float* __restrict__ cache,
                            int* __restrict__ kv_id) {
  if (threadIdx.x != 0 || blockIdx.x != 0) return;
  float mags[CHKV];
  for (int i = 0; i < CHKV; ++i) {
    float s = 0.0f;
    for (int b = 0; b < CB; ++b) s += sqrtf(ksumsq[b * CHKV + i]);
    mags[i] = s;
  }
  float diff[CHKV], ds = 0.0f;
  for (int i = 0; i < CHKV; ++i) { diff[i] = fabsf(cache[i] - mags[i]); ds += diff[i]; }
  int r[CHKV], rs = 0;
  for (int i = 0; i < CHKV; ++i) { r[i] = (int)rintf(diff[i] / ds * CH); rs += r[i]; }
  while (rs > CH) {                     // r[argmax] -= 1
    int am = 0;
    for (int i = 1; i < CHKV; ++i) if (r[i] > r[am]) am = i;
    r[am]--; rs--;
  }
  while (rs < CH) {                     // r[argmin] += 1
    int am = 0;
    for (int i = 1; i < CHKV; ++i) if (r[i] < r[am]) am = i;
    r[am]++; rs++;
  }
  int cums[CHKV], c = 0;
  for (int i = 0; i < CHKV; ++i) { c += r[i]; cums[i] = c; }
  for (int h = 0; h < CH; ++h) {        // searchsorted(cums, h, 'right')
    int id = 0;
    while (id < CHKV && cums[id] <= h) id++;
    kv_id[h] = id < CHKV ? id : CHKV - 1;
  }
}

// ---------------------------------------------------------------------------
// Flash attention, one wave per (b, h, 16-query tile).
// Computes S^T = K*Q^T so softmax stats are per-lane scalars, then
// O^T = V^T * P^T with P^T repacked to a B-fragment entirely in registers.
// ---------------------------------------------------------------------------
__global__ __launch_bounds__(32) void dgqa_attn(
    const unsigned short* __restrict__ qb,  // (B*P) x 1024, q pre-scaled
    const unsigned short* __restrict__ kb,  // (B*P) x 512
    const unsigned short* __restrict__ vT,  // (B*HKV*64) x P
    const int* __restrict__ kv_id,
    unsigned short* __restrict__ ob) {      // (B*P) x 1024
  int b = blockIdx.z, h = blockIdx.y;
  int q0 = blockIdx.x * 16;
  int kvh = kv_id[h];
  int lane = threadIdx.x & 31;
  int sel = lane >> 4;

  const unsigned short* qbase = qb + (size_t)b * CP * CDIM + h * CD;
  const unsigned short* kbase = kb + (size_t)b * CP * CKV + kvh * CD;
  const unsigned short* vbase = vT + (size_t)(b * CHKV + kvh) * CD * CP;

  // Q^T as B-fragments (N = 16 queries, K = D = 64 -> two chunks), loaded once.
  v16bf Bq0 = load_frag16(qbase, q0, CDIM, 0);
  v16bf Bq1 = load_frag16(qbase, q0, CDIM, 32);

  v8f O[4];
#pragma unroll
  for (int df = 0; df < 4; ++df)
#pragma unroll
    for (int r = 0; r < 8; ++r) O[df][r] = 0.0f;
  float m = -INFINITY, l = 0.0f;

  for (int kt = 0; kt < CP; kt += 32) {
    // Prefetch next K/V tiles while this one computes (global_prefetch_b8).
    if (kt + 32 < CP) {
      __builtin_prefetch((const void*)(kbase + (size_t)(kt + 32) * CKV), 0, 1);
      __builtin_prefetch((const void*)(vbase + kt + 32), 0, 1);
    }

    // K tile as A-fragments (rows = keys)
    v16bf a00 = load_frag16(kbase, kt,      CKV, 0);
    v16bf a01 = load_frag16(kbase, kt,      CKV, 32);
    v16bf a10 = load_frag16(kbase, kt + 16, CKV, 0);
    v16bf a11 = load_frag16(kbase, kt + 16, CKV, 32);

    v8f S0, S1;
#pragma unroll
    for (int r = 0; r < 8; ++r) { S0[r] = 0.0f; S1[r] = 0.0f; }
    S0 = wmma_bf16(a00, Bq0, S0);
    S0 = wmma_bf16(a01, Bq1, S0);
    S1 = wmma_bf16(a10, Bq0, S1);
    S1 = wmma_bf16(a11, Bq1, S1);

    // Online softmax over this tile's 32 keys; per-lane q column.
    float smax = -INFINITY;
#pragma unroll
    for (int r = 0; r < 8; ++r) smax = fmaxf(smax, fmaxf(S0[r], S1[r]));
    smax = fmaxf(smax, __shfl_xor(smax, 16, 32));
    float mnew = fmaxf(m, smax);
    float corr = __expf(m - mnew);
    float psum = 0.0f;
#pragma unroll
    for (int r = 0; r < 8; ++r) {
      float p0 = __expf(S0[r] - mnew);
      float p1 = __expf(S1[r] - mnew);
      S0[r] = p0; S1[r] = p1;
      psum += p0 + p1;
    }
    psum += __shfl_xor(psum, 16, 32);
    l = l * corr + psum;
    m = mnew;
#pragma unroll
    for (int df = 0; df < 4; ++df)
#pragma unroll
      for (int r = 0; r < 8; ++r) O[df][r] *= corr;

    // Repack P^T (C layout) -> B-fragment bf16 pairs, purely in-lane.
    union { unsigned u[8]; v16bf v; } bp;
#pragma unroll
    for (int v2 = 0; v2 < 4; ++v2) {
      bp.u[v2]     = (unsigned)f2bf(S0[2 * v2]) |
                     ((unsigned)f2bf(S0[2 * v2 + 1]) << 16);
      bp.u[v2 + 4] = (unsigned)f2bf(S1[2 * v2]) |
                     ((unsigned)f2bf(S1[2 * v2 + 1]) << 16);
    }

    // O^T += V^T * P^T
#pragma unroll
    for (int df = 0; df < 4; ++df) {
      v16bf av = load_frag16(vbase, df * 16, CP, kt);
      O[df] = wmma_bf16(av, bp.v, O[df]);
    }
  }

  float invl = 1.0f / l;
  int qcol = lane & 15;
  size_t orow = ((size_t)b * CP + q0 + qcol) * CDIM + h * CD;
#pragma unroll
  for (int df = 0; df < 4; ++df) {
    union { uint4 q; unsigned short s[8]; } pk;
#pragma unroll
    for (int r = 0; r < 8; ++r) pk.s[r] = f2bf(O[df][r] * invl);
    *(uint4*)(ob + orow + df * 16 + sel * 8) = pk.q;
  }
}

// ---------------------------------------------------------------------------
extern "C" void kernel_launch(void* const* d_in, const int* in_sizes, int n_in,
                              void* d_out, int out_size, void* d_ws,
                              size_t ws_size, hipStream_t stream) {
  (void)in_sizes; (void)n_in; (void)out_size; (void)ws_size;
  const float* x     = (const float*)d_in[0];
  const float* Wq    = (const float*)d_in[1];
  const float* Wk    = (const float*)d_in[2];
  const float* Wv    = (const float*)d_in[3];
  const float* Wp    = (const float*)d_in[4];
  const float* bp    = (const float*)d_in[5];
  const float* cache = (const float*)d_in[6];
  float* out = (float*)d_out;

  char* ws = (char*)d_ws;
  size_t off = 0;
  auto alloc = [&](size_t bytes) -> void* {
    void* p = ws + off;
    off += (bytes + 255) & ~(size_t)255;
    return p;
  };
  const int M = CB * CP;  // 8192
  unsigned short* xb  = (unsigned short*)alloc((size_t)M * CDIM * 2);
  unsigned short* Wqt = (unsigned short*)alloc((size_t)CDIM * CDIM * 2);
  unsigned short* Wkt = (unsigned short*)alloc((size_t)CKV * CDIM * 2);
  unsigned short* Wvt = (unsigned short*)alloc((size_t)CKV * CDIM * 2);
  unsigned short* Wpt = (unsigned short*)alloc((size_t)CDIM * CDIM * 2);
  unsigned short* qbuf = (unsigned short*)alloc((size_t)M * CDIM * 2);
  unsigned short* kbuf = (unsigned short*)alloc((size_t)M * CKV * 2);
  unsigned short* vbuf = (unsigned short*)alloc((size_t)M * CKV * 2);
  unsigned short* vT   = (unsigned short*)alloc((size_t)M * CKV * 2);
  unsigned short* obuf = (unsigned short*)alloc((size_t)M * CDIM * 2);
  float* ksq  = (float*)alloc(CB * CHKV * sizeof(float));
  int*   kvid = (int*)alloc(CH * sizeof(int));

  // 1) fp32 -> bf16 conversions (+ weight transposes to [N][K])
  dgqa_cvt_bf16<<<2048, 256, 0, stream>>>(x, xb, M * CDIM);
  dgqa_transpose_cvt<<<2048, 256, 0, stream>>>(Wq, Wqt, CDIM, CDIM);
  dgqa_transpose_cvt<<<2048, 256, 0, stream>>>(Wk, Wkt, CDIM, CKV);
  dgqa_transpose_cvt<<<2048, 256, 0, stream>>>(Wv, Wvt, CDIM, CKV);
  dgqa_transpose_cvt<<<2048, 256, 0, stream>>>(Wp, Wpt, CDIM, CDIM);

  // 2) projections (q scaled by D^-0.5 = 0.125 inside the GEMM)
  dim3 gq(CDIM / 256, M / 128);
  dgqa_gemm_bf16<<<gq, 256, 0, stream>>>(xb, Wqt, M, CDIM, CDIM, 0.125f,
                                         nullptr, qbuf, nullptr);
  dim3 gkv(CKV / 256, M / 128);
  dgqa_gemm_bf16<<<gkv, 256, 0, stream>>>(xb, Wkt, M, CKV, CDIM, 1.0f,
                                          nullptr, kbuf, nullptr);
  dgqa_gemm_bf16<<<gkv, 256, 0, stream>>>(xb, Wvt, M, CKV, CDIM, 1.0f,
                                          nullptr, vbuf, nullptr);

  // 3) V transpose for the PV matmul; head-routing ratios
  dgqa_transpose_v<<<2048, 256, 0, stream>>>(vbuf, vT);
  dgqa_ksumsq<<<CB * CHKV, 256, 0, stream>>>(kbuf, ksq);
  dgqa_ratios<<<1, 1, 0, stream>>>(ksq, cache, kvid);

  // 4) flash attention: one wave per (b, h, 16-query tile)
  dim3 ga(CP / 16, CH, CB);
  dgqa_attn<<<ga, 32, 0, stream>>>(qbuf, kbuf, vT, kvid, obuf);

  // 5) output projection + bias, f32 result
  dim3 gp(CDIM / 256, M / 128);
  dgqa_gemm_bf16<<<gp, 256, 0, stream>>>(obuf, Wpt, M, CDIM, CDIM, 1.0f, bp,
                                         nullptr, out);
}
